// TwoDimPlanesModel_53094385713380
// MI455X (gfx1250) — compile-verified, matrix operand order checked by reading, not run on hardware
//
#include <hip/hip_runtime.h>
#include <hip/hip_bf16.h>

typedef __attribute__((ext_vector_type(16))) _Float16 v16h;
typedef __attribute__((ext_vector_type(8)))  float    v8f;

#define RESN 512
#define PLSZ (512*512)

// Swizzled f16 weight buffer layout (halves):
//   [     0,  8192) dW0   : MB=8, KB=2 (K padded 48->64)
//   [  8192, 57344) dWh[3]: MB=8, KB=4, 16384 halves each
//   [ 57344, 65536) rW0   : MB=8, KB=2
//   [ 65536,114688) rWh[3]: MB=8, KB=4
// then f32 aux region (1536 floats = 3072 halves):
//   [0,128) d_b0 | [128,512) d_bh | [512,640) r_b0 | [640,1024) r_bh
//   [1024,1152) alpha_w | [1152,1536) rgb_w ([128,3] row-major)
// Fragment (mb,kb): 512 halves = 32 lanes x 16 halves, per-lane contiguous.
// A-fragment element mapping (16-bit A 16x32, wave32):
//   m = mb*16 + (lane&15)
//   k = kb*32 + (h&7) + (h>>3)*16 + (lane>=16 ? 8 : 0)

#define WTOTAL_HALVES 114688
#define WTOTAL_BYTES  229376
#define AUX_FLOATS    1536
#define AUX_BYTES     6144
#define COPY_V4       ((WTOTAL_BYTES + AUX_BYTES) / 16)   // 14720
#define SMEM_BYTES    (WTOTAL_BYTES + AUX_BYTES)          // 235520

__global__ void swizzle_weights(const float* __restrict__ d_w0,
                                const float* __restrict__ d_wh,
                                const float* __restrict__ r_w0,
                                const float* __restrict__ r_wh,
                                _Float16* __restrict__ wout) {
    int tid = blockIdx.x * blockDim.x + threadIdx.x;
    if (tid >= WTOTAL_HALVES) return;
    const float* src; int KB, Kreal, local;
    if (tid < 8192)       { src = d_w0;  KB = 2; Kreal = 48;  local = tid; }
    else if (tid < 57344) { int t = tid - 8192;  src = d_wh + (t / 16384) * 16384;
                            KB = 4; Kreal = 128; local = t & 16383; }
    else if (tid < 65536) { src = r_w0;  KB = 2; Kreal = 48;  local = tid - 57344; }
    else                  { int t = tid - 65536; src = r_wh + (t / 16384) * 16384;
                            KB = 4; Kreal = 128; local = t & 16383; }
    int frag = local >> 9;
    int r    = local & 511;
    int lane = r >> 4;
    int h    = r & 15;
    int mb = frag / KB, kb = frag % KB;
    int m   = mb * 16 + (lane & 15);
    int off = (lane >= 16) ? 8 : 0;
    int k   = kb * 32 + (h & 7) + ((h >> 3) << 4) + off;
    float v = (k < Kreal) ? src[k * 128 + m] : 0.0f;
    wout[tid] = (_Float16)v;
}

__global__ void pack_aux(const float* __restrict__ d_b0, const float* __restrict__ d_bh,
                         const float* __restrict__ r_b0, const float* __restrict__ r_bh,
                         const float* __restrict__ alpha_w, const float* __restrict__ rgb_w,
                         float* __restrict__ dst) {
    int i = blockIdx.x * blockDim.x + threadIdx.x;
    if (i >= AUX_FLOATS) return;
    float v;
    if (i < 128)       v = d_b0[i];
    else if (i < 512)  v = d_bh[i - 128];
    else if (i < 640)  v = r_b0[i - 512];
    else if (i < 1024) v = r_bh[i - 640];
    else if (i < 1152) v = alpha_w[i - 1024];
    else               v = rgb_w[i - 1152];
    dst[i] = v;
}

struct Samp { int i00, i01, i10, i11; float w00, w01, w10, w11; };

__device__ __forceinline__ Samp make_samp(float gx, float gy) {
    float ix = fminf(fmaxf((gx + 1.0f) * 0.5f * 511.0f, 0.0f), 511.0f);
    float iy = fminf(fmaxf((gy + 1.0f) * 0.5f * 511.0f, 0.0f), 511.0f);
    float xf = floorf(ix), yf = floorf(iy);
    float wx = ix - xf, wy = iy - yf;
    int x0 = (int)xf, y0 = (int)yf;
    int x1 = min(x0 + 1, 511), y1 = min(y0 + 1, 511);
    Samp s;
    s.i00 = y0 * RESN + x0; s.i01 = y0 * RESN + x1;
    s.i10 = y1 * RESN + x0; s.i11 = y1 * RESN + x1;
    float owx = 1.0f - wx, owy = 1.0f - wy;
    s.w00 = owx * owy; s.w01 = wx * owy; s.w10 = owx * wy; s.w11 = wx * wy;
    return s;
}

__device__ __forceinline__ v8f wmma_acc(v16h a, v16h b, v8f c) {
    return __builtin_amdgcn_wmma_f32_16x16x32_f16(false, a, false, b, (short)0, c,
                                                  false, false);
}

__device__ __forceinline__ v16h afrag(const _Float16* __restrict__ smem,
                                      int base, int frag, int lane) {
    return *(const v16h*)(smem + base + frag * 512 + lane * 16);
}

__device__ __forceinline__ v8f bias8(const float* __restrict__ bias, int t, int off) {
    return *(const v8f*)(bias + t * 16 + off);   // 32B aligned (off in {0,8})
}

// pack two f32 tiles to one f16 fragment, then packed-f16 ReLU
// (cvt_pk output is canonical -> v_pk_max_num_f16 without canonicalize)
__device__ __forceinline__ v16h pack_relu(v8f c0, v8f c1) {
    v16h b;
    #pragma unroll
    for (int h = 0; h < 8; ++h) {
        b[h]     = (_Float16)c0[h];
        b[h + 8] = (_Float16)c1[h];
    }
    v16h z = {};
    return __builtin_elementwise_max(b, z);
}

// Fused hidden layer: C initialized with bias (from LDS), so D = W*x + b directly.
// Per tile-pair: matmul then f16 pack + packed ReLU. bin/bout must not alias.
template<int KB>
__device__ __forceinline__ void layer_fused(const _Float16* __restrict__ smem, int base,
                                            const float* __restrict__ bias,
                                            int off, int lane,
                                            const v16h* bin, v16h* bout) {
    #pragma unroll
    for (int t = 0; t < 4; ++t) {
        v8f c0 = bias8(bias, 2 * t, off);
        v8f c1 = bias8(bias, 2 * t + 1, off);
        #pragma unroll
        for (int kb = 0; kb < KB; ++kb)
            c0 = wmma_acc(afrag(smem, base, (2 * t) * KB + kb, lane), bin[kb], c0);
        #pragma unroll
        for (int kb = 0; kb < KB; ++kb)
            c1 = wmma_acc(afrag(smem, base, (2 * t + 1) * KB + kb, lane), bin[kb], c1);
        bout[t] = pack_relu(c0, c1);
    }
}

// Final hidden layer fused with a 1-output head: returns this lane's partial dot.
template<int KB>
__device__ __forceinline__ float layer_head1(const _Float16* __restrict__ smem, int base,
                                             const float* __restrict__ bias,
                                             const float* __restrict__ hw,
                                             int off, int lane, const v16h* bin) {
    float pa = 0.0f;
    #pragma unroll
    for (int t = 0; t < 4; ++t) {
        v8f c0 = bias8(bias, 2 * t, off);
        v8f c1 = bias8(bias, 2 * t + 1, off);
        #pragma unroll
        for (int kb = 0; kb < KB; ++kb)
            c0 = wmma_acc(afrag(smem, base, (2 * t) * KB + kb, lane), bin[kb], c0);
        #pragma unroll
        for (int kb = 0; kb < KB; ++kb)
            c1 = wmma_acc(afrag(smem, base, (2 * t + 1) * KB + kb, lane), bin[kb], c1);
        const float* w0 = hw + (2 * t) * 16 + off;
        const float* w1 = hw + (2 * t + 1) * 16 + off;
        #pragma unroll
        for (int h = 0; h < 8; ++h) {
            pa += fmaxf(c0[h], 0.0f) * w0[h];
            pa += fmaxf(c1[h], 0.0f) * w1[h];
        }
    }
    return pa;
}

// Final hidden layer fused with a 3-output head (rgb_w is [128,3] row-major).
template<int KB>
__device__ __forceinline__ void layer_head3(const _Float16* __restrict__ smem, int base,
                                            const float* __restrict__ bias,
                                            const float* __restrict__ hw,
                                            int off, int lane, const v16h* bin,
                                            float pr[3]) {
    #pragma unroll
    for (int t = 0; t < 4; ++t) {
        v8f c0 = bias8(bias, 2 * t, off);
        v8f c1 = bias8(bias, 2 * t + 1, off);
        #pragma unroll
        for (int kb = 0; kb < KB; ++kb)
            c0 = wmma_acc(afrag(smem, base, (2 * t) * KB + kb, lane), bin[kb], c0);
        #pragma unroll
        for (int kb = 0; kb < KB; ++kb)
            c1 = wmma_acc(afrag(smem, base, (2 * t + 1) * KB + kb, lane), bin[kb], c1);
        #pragma unroll
        for (int h = 0; h < 8; ++h) {
            int f0 = (2 * t) * 16 + off + h;
            int f1 = f0 + 16;
            float a0 = fmaxf(c0[h], 0.0f);
            float a1 = fmaxf(c1[h], 0.0f);
            const float* w0 = hw + f0 * 3;
            const float* w1 = hw + f1 * 3;
            pr[0] += a0 * w0[0] + a1 * w1[0];
            pr[1] += a0 * w0[1] + a1 * w1[1];
            pr[2] += a0 * w0[2] + a1 * w1[2];
        }
    }
}

__global__ __launch_bounds__(512)
void triplane_kernel(const float* __restrict__ x,
                     const float* __restrict__ planes,
                     const _Float16* __restrict__ wf,
                     const float* __restrict__ alpha_b,
                     const float* __restrict__ rgb_b,
                     float* __restrict__ out) {
    extern __shared__ _Float16 smem[];
    // [0,229376): weights | [229376,235520): f32 aux

    // ---- async-stage weights + aux into LDS (CDNA5 async copy) ----
    {
        #pragma unroll 4
        for (int i = threadIdx.x; i < COPY_V4; i += 512) {
            unsigned int ldst =
                (unsigned int)(unsigned long long)(uintptr_t)(smem) + (unsigned int)(i * 16);
            unsigned long long ga =
                (unsigned long long)(uintptr_t)wf + (unsigned long long)i * 16ull;
            asm volatile("global_load_async_to_lds_b128 %0, %1, off"
                         :: "v"(ldst), "v"(ga) : "memory");
        }
        asm volatile("s_wait_asynccnt 0" ::: "memory");
        __syncthreads();
    }

    const float* fb     = (const float*)(smem + WTOTAL_HALVES);
    const float* L_db0  = fb;
    const float* L_dbh  = fb + 128;
    const float* L_rb0  = fb + 512;
    const float* L_rbh  = fb + 640;
    const float* L_aw   = fb + 1024;
    const float* L_rgbw = fb + 1152;

    const int lane = threadIdx.x & 31;
    const int wave = threadIdx.x >> 5;
    const int p    = blockIdx.x * 256 + wave * 16 + (lane & 15);
    const int off  = (lane & 16) ? 8 : 0;

    // ---- coordinates ----
    const float* xp = x + (size_t)p * 6;
    float px = xp[0], py = xp[1], pz = xp[2];
    float dx = xp[3], dy = xp[4], dz = xp[5];
    float az = atan2f(dy, dx);
    float el = atan2f(dz, sqrtf(dx * dx + dy * dy));
    const float INV_PI = 0.3183098861837907f;
    float n0 = 2.0f * px - 1.0f, n1 = 2.0f * py - 1.0f, n2 = 2.0f * pz - 1.0f;
    float n3 = az * INV_PI, n4 = el * (2.0f * INV_PI);

    Samp ss[3];
    ss[0] = make_samp(n1, n2);
    ss[1] = make_samp(n0, n2);
    ss[2] = make_samp(n0, n1);
    Samp s3 = make_samp(n3, n4);

    // ---- grid sample: lane handles channels {g*16+off+j : g<3, j<8} of its point
    float proj[24];
    float pv[24];
    #pragma unroll
    for (int i = 0; i < 24; ++i) proj[i] = 0.0f;

    #pragma unroll
    for (int pl = 0; pl < 3; ++pl) {
        const Samp s = ss[pl];
        const float* Pb = planes + (size_t)pl * 48 * PLSZ;
        #pragma unroll
        for (int g = 0; g < 3; ++g) {
            #pragma unroll
            for (int j = 0; j < 8; ++j) {
                const float* P = Pb + (size_t)(g * 16 + off + j) * PLSZ;
                proj[g * 8 + j] += P[s.i00] * s.w00 + P[s.i01] * s.w01 +
                                   P[s.i10] * s.w10 + P[s.i11] * s.w11;
            }
        }
    }
    {
        const float* Pb = planes + (size_t)3 * 48 * PLSZ;
        #pragma unroll
        for (int g = 0; g < 3; ++g) {
            #pragma unroll
            for (int j = 0; j < 8; ++j) {
                const float* P = Pb + (size_t)(g * 16 + off + j) * PLSZ;
                pv[g * 8 + j] = P[s3.i00] * s3.w00 + P[s3.i01] * s3.w01 +
                                P[s3.i10] * s3.w10 + P[s3.i11] * s3.w11;
            }
        }
    }

    // ---- layer-0 B fragments (K padded to 64: kb=1 upper half is zero)
    v16h bD[2], bR[2];
    #pragma unroll
    for (int h = 0; h < 8; ++h) {
        bD[0][h]     = (_Float16)proj[h];
        bD[0][h + 8] = (_Float16)proj[8 + h];
        bD[1][h]     = (_Float16)proj[16 + h];
        bD[1][h + 8] = (_Float16)0.0f;
        bR[0][h]     = (_Float16)(proj[h] + pv[h]);
        bR[0][h + 8] = (_Float16)(proj[8 + h] + pv[8 + h]);
        bR[1][h]     = (_Float16)(proj[16 + h] + pv[16 + h]);
        bR[1][h + 8] = (_Float16)0.0f;
    }

    __builtin_amdgcn_sched_barrier(0);   // keep sampling out of the MLP region

    v16h fA[4], fB[4];   // ping-pong activation fragments

    // ---- density MLP ----
    layer_fused<2>(smem, 0,               L_db0,         off, lane, bD, fA);
    layer_fused<4>(smem, 8192 + 0*16384,  L_dbh + 0*128, off, lane, fA, fB);
    layer_fused<4>(smem, 8192 + 1*16384,  L_dbh + 1*128, off, lane, fB, fA);
    float pa = layer_head1<4>(smem, 8192 + 2*16384, L_dbh + 2*128, L_aw,
                              off, lane, fA);
    pa += __shfl_xor(pa, 16, 32);
    float alpha = pa + alpha_b[0];

    __builtin_amdgcn_sched_barrier(0);

    // ---- rgb MLP ----
    layer_fused<2>(smem, 57344,           L_rb0,         off, lane, bR, fA);
    layer_fused<4>(smem, 65536 + 0*16384, L_rbh + 0*128, off, lane, fA, fB);
    layer_fused<4>(smem, 65536 + 1*16384, L_rbh + 1*128, off, lane, fB, fA);
    float pr[3] = {0.0f, 0.0f, 0.0f};
    layer_head3<4>(smem, 65536 + 2*16384, L_rbh + 2*128, L_rgbw, off, lane, fA, pr);
    pr[0] += __shfl_xor(pr[0], 16, 32);
    pr[1] += __shfl_xor(pr[1], 16, 32);
    pr[2] += __shfl_xor(pr[2], 16, 32);

    if (lane < 16) {
        float4 o;
        o.x = pr[0] + rgb_b[0];
        o.y = pr[1] + rgb_b[1];
        o.z = pr[2] + rgb_b[2];
        o.w = alpha;
        *(float4*)(out + (size_t)p * 4) = o;
    }
}

extern "C" void kernel_launch(void* const* d_in, const int* in_sizes, int n_in,
                              void* d_out, int out_size, void* d_ws, size_t ws_size,
                              hipStream_t stream) {
    const float* x       = (const float*)d_in[0];
    const float* planes  = (const float*)d_in[1];
    const float* d_w0    = (const float*)d_in[2];
    const float* d_b0    = (const float*)d_in[3];
    const float* d_wh    = (const float*)d_in[4];
    const float* d_bh    = (const float*)d_in[5];
    const float* alpha_w = (const float*)d_in[6];
    const float* alpha_b = (const float*)d_in[7];
    const float* r_w0    = (const float*)d_in[8];
    const float* r_b0    = (const float*)d_in[9];
    const float* r_wh    = (const float*)d_in[10];
    const float* r_bh    = (const float*)d_in[11];
    const float* rgb_w   = (const float*)d_in[12];
    const float* rgb_b   = (const float*)d_in[13];
    float* out = (float*)d_out;
    _Float16* wf = (_Float16*)d_ws;   // 235520 bytes used

    (void)hipFuncSetAttribute((const void*)triplane_kernel,
                              hipFuncAttributeMaxDynamicSharedMemorySize, SMEM_BYTES);

    hipLaunchKernelGGL(swizzle_weights, dim3((WTOTAL_HALVES + 255) / 256), dim3(256), 0, stream,
                       d_w0, d_wh, r_w0, r_wh, wf);
    hipLaunchKernelGGL(pack_aux, dim3((AUX_FLOATS + 255) / 256), dim3(256), 0, stream,
                       d_b0, d_bh, r_b0, r_bh, alpha_w, rgb_w,
                       (float*)(wf + WTOTAL_HALVES));

    const int nblocks = 524288 / 256;  // 16 waves x 16 points per block
    hipLaunchKernelGGL(triplane_kernel, dim3(nblocks), dim3(512), SMEM_BYTES, stream,
                       x, planes, (const _Float16*)wf, alpha_b, rgb_b, out);
    (void)in_sizes; (void)n_in; (void)out_size; (void)ws_size;
}
